// GCN_48687749268141
// MI455X (gfx1250) — matrix operand order repaired and checked by reading, not compiled, not run on hardware
//
#include <hip/hip_runtime.h>
#include <hip/hip_bf16.h>

// GCN forward for MI455X (gfx1250, wave32).
//  - Laplacian build: elementwise + row reductions (HBM-bound, negligible).
//  - eigh: persistent cyclic-Jacobi eigensolver (fp32), round-robin pairing,
//    device-wide atomic barrier; S+V (33.5MB) are L2-resident (192MB L2).
//  - GraphConv / MLP GEMMs: bf16 operands, f32 accumulate via
//    v_wmma_f32_16x16x32_bf16.
//      * A tiles: Tensor Data Mover DMA into double-buffered LDS
//        (tensor_load_to_lds, TENSORcnt), fragments = 2x ds_load_b128.
//      * B tiles: software-pipelined global_load_b128 -> registers during
//        WMMA compute, then scattered into a fragment-major swizzled LDS
//        layout so each B fragment is one contiguous 32B LDS read.
//  - pool + classifier: tiny fp32 kernels.

typedef __attribute__((ext_vector_type(16))) __bf16 v16bf;
typedef __attribute__((ext_vector_type(8)))  __bf16 v8bf;
typedef __attribute__((ext_vector_type(8)))  float  v8f;
typedef unsigned int u32x4 __attribute__((ext_vector_type(4)));
typedef int          i32x8 __attribute__((ext_vector_type(8)));
typedef int          i32x4 __attribute__((ext_vector_type(4)));

constexpr int N     = 2048;
constexpr int Bn    = 64;
constexpr int Din   = 128;
constexpr int Hd    = 128;
constexpr int OUTC  = 7;
constexpr int NPAIR = N / 2;

constexpr int JAC_BLOCKS = 96;
constexpr int NSWEEPS    = 6;

// ---------------- workspace layout (bytes, all offsets 256B aligned) --------
constexpr size_t SZ_NN_F  = (size_t)N * N * sizeof(float);     // 16 MiB
constexpr size_t SZ_NN_BF = (size_t)N * N * sizeof(__bf16);    // 8 MiB
constexpr size_t SZ_W_BF  = (size_t)128 * 128 * sizeof(__bf16);
constexpr size_t SZ_ACT   = (size_t)Bn * N * Hd * sizeof(__bf16); // 32 MiB

constexpr size_t OFF_S    = 0;
constexpr size_t OFF_V    = OFF_S   + SZ_NN_F;
constexpr size_t OFF_DEG  = OFF_V   + SZ_NN_F;
constexpr size_t OFF_DIS  = OFF_DEG + (size_t)N * sizeof(float);
constexpr size_t OFF_CS   = OFF_DIS + (size_t)N * sizeof(float);
constexpr size_t OFF_BAR  = OFF_CS  + (size_t)2 * NPAIR * sizeof(float);
constexpr size_t OFF_UBF  = OFF_BAR + 256;
constexpr size_t OFF_UBT  = OFF_UBF + SZ_NN_BF;
constexpr size_t OFF_WT0  = OFF_UBT + SZ_NN_BF;
constexpr size_t OFF_WT1  = OFF_WT0 + SZ_W_BF;
constexpr size_t OFF_WT2  = OFF_WT1 + SZ_W_BF;
constexpr size_t OFF_WT3  = OFF_WT2 + SZ_W_BF;
constexpr size_t OFF_BUF0 = OFF_WT3 + SZ_W_BF;
constexpr size_t OFF_BUF1 = OFF_BUF0 + SZ_ACT;
constexpr size_t OFF_POOL = OFF_BUF1 + SZ_ACT;

// ---------------- Laplacian build ------------------------------------------
__global__ __launch_bounds__(256) void k_deg(const float* __restrict__ adj,
                                             float* __restrict__ deg,
                                             float* __restrict__ dis) {
  const int i = blockIdx.x;
  __shared__ float red[256];
  float s = 0.f;
  for (int j = threadIdx.x; j < N; j += 256) {
    float v = adj[(size_t)i * N + j];
    s += (v > 0.f) ? v : 0.f;
  }
  red[threadIdx.x] = s;
  __syncthreads();
  for (int o = 128; o > 0; o >>= 1) {
    if ((int)threadIdx.x < o) red[threadIdx.x] += red[threadIdx.x + o];
    __syncthreads();
  }
  if (threadIdx.x == 0) {
    deg[i] = red[0];
    dis[i] = rsqrtf(red[0]);   // deg ** -0.5
  }
}

__global__ __launch_bounds__(256) void k_lsym(const float* __restrict__ adj,
                                              const float* __restrict__ deg,
                                              const float* __restrict__ dis,
                                              float* __restrict__ S) {
  const int i = blockIdx.x;
  const float di = dis[i], dgi = deg[i];
  for (int j = threadIdx.x; j < N; j += 256) {
    float a = adj[(size_t)i * N + j];
    a = (a > 0.f) ? a : 0.f;
    const float L = ((i == j) ? dgi : 0.f) - a;
    S[(size_t)i * N + j] = di * L * dis[j];   // symmetric by construction
  }
}

// ---------------- Jacobi eigensolver ---------------------------------------
__global__ void k_init_bar(unsigned* bar) { bar[0] = 0u; bar[1] = 0u; }

__device__ __forceinline__ int rr_idx(int m, int r) {
  // round-robin tournament permutation: element 0 fixed, others rotate
  return (m == 0) ? 0 : 1 + ((m - 1 + r) % (N - 1));
}

__device__ __forceinline__ void grid_sync(unsigned* bar) {
  volatile unsigned* vb = bar;
  __syncthreads();
  if (threadIdx.x == 0) {
    __threadfence();
    const unsigned gen = vb[1];
    if (atomicAdd(&bar[0], 1u) == gridDim.x - 1) {
      vb[0] = 0u;
      __threadfence();
      atomicAdd(&bar[1], 1u);
    } else {
      while (vb[1] == gen) __builtin_amdgcn_s_sleep(1);
    }
    __threadfence();
  }
  __syncthreads();
}

__global__ __launch_bounds__(256)
void k_jacobi(float* __restrict__ S, float* __restrict__ V,
              float* __restrict__ cs, unsigned* __restrict__ bar) {
  const int tid = blockIdx.x * blockDim.x + threadIdx.x;
  const int nth = gridDim.x * blockDim.x;

  // V = I
  for (int idx = tid; idx < N * N; idx += nth)
    V[idx] = ((idx / N) == (idx % N)) ? 1.f : 0.f;
  grid_sync(bar);

  for (int sw = 0; sw < NSWEEPS; ++sw) {
    for (int step = 0; step < N - 1; ++step) {
      // phase A: rotation angles for 1024 disjoint pairs
      if (tid < NPAIR) {
        int p = rr_idx(tid, step), q = rr_idx(N - 1 - tid, step);
        if (p > q) { int t = p; p = q; q = t; }
        const float app = S[(size_t)p * N + p];
        const float aqq = S[(size_t)q * N + q];
        const float apq = S[(size_t)p * N + q];
        float c, s;
        if (fabsf(apq) > 1e-30f) {
          const float th = (aqq - app) / (2.f * apq);
          const float t  = copysignf(1.f, th) / (fabsf(th) + sqrtf(th * th + 1.f));
          c = rsqrtf(t * t + 1.f);
          s = t * c;
        } else { c = 1.f; s = 0.f; }
        cs[2 * tid]     = c;
        cs[2 * tid + 1] = s;
      }
      grid_sync(bar);

      // phase B: column rotations on S and V (pairs own disjoint columns)
      for (int w = tid; w < N * NPAIR; w += nth) {
        const int k = w / N, i = w % N;
        int p = rr_idx(k, step), q = rr_idx(N - 1 - k, step);
        if (p > q) { int t = p; p = q; q = t; }
        const float c = cs[2 * k], s = cs[2 * k + 1];
        const size_t ip = (size_t)i * N + p, iq = (size_t)i * N + q;
        const float sp = S[ip], sq = S[iq];
        S[ip] = c * sp - s * sq;
        S[iq] = s * sp + c * sq;
        const float vp = V[ip], vq = V[iq];
        V[ip] = c * vp - s * vq;
        V[iq] = s * vp + c * vq;
      }
      grid_sync(bar);

      // phase C: row rotations on S
      for (int w = tid; w < N * NPAIR; w += nth) {
        const int k = w / N, j = w % N;
        int p = rr_idx(k, step), q = rr_idx(N - 1 - k, step);
        if (p > q) { int t = p; p = q; q = t; }
        const float c = cs[2 * k], s = cs[2 * k + 1];
        const size_t pj = (size_t)p * N + j, qj = (size_t)q * N + j;
        const float sp = S[pj], sq = S[qj];
        S[pj] = c * sp - s * sq;
        S[qj] = s * sp + c * sq;
      }
      grid_sync(bar);
    }
  }
}

// ---------------- fp32 -> bf16 conversions ---------------------------------
__global__ __launch_bounds__(256) void k_cvtU(const float* __restrict__ V,
                                              __bf16* __restrict__ U,
                                              __bf16* __restrict__ Ut) {
  const int i = blockIdx.x;
  for (int j = threadIdx.x; j < N; j += 256) {
    const __bf16 b = (__bf16)V[(size_t)i * N + j];
    U [(size_t)i * N + j] = b;
    Ut[(size_t)j * N + i] = b;
  }
}

__global__ __launch_bounds__(256) void k_cvt(const float* __restrict__ src,
                                             __bf16* __restrict__ dst,
                                             long long n) {
  long long i = (long long)blockIdx.x * blockDim.x + threadIdx.x;
  const long long st = (long long)gridDim.x * blockDim.x;
  for (; i < n; i += st) dst[i] = (__bf16)src[i];
}

__global__ __launch_bounds__(128) void k_cvtWT(const float* __restrict__ W,
                                               __bf16* __restrict__ Wt) {
  const int o = blockIdx.x, i = threadIdx.x;        // W: [128,128] torch layout
  Wt[i * 128 + o] = (__bf16)W[o * 128 + i];         // Wt[k][n] = W[n][k]
}

// ---------------- Tensor Data Mover: 2D tile -> LDS ------------------------
// Builds a D# per CDNA5 ISA §8 and issues TENSOR_LOAD_TO_LDS (TENSORcnt).
// group0: count=1 | lds_addr | global_addr[56:0] | type=2 ("image")
// group1: data_size=1(2B) | tensor_dim0/1 | tile_dim0/1 | tensor_dim0_stride
__device__ __forceinline__ unsigned lds_off(const void* p) {
  // lower 32 bits of a flat-cast LDS pointer = byte offset in wave's LDS
  return (unsigned)(unsigned long long)p;
}

__device__ __forceinline__ void tdm_load_2d(unsigned lds_addr, const void* gp,
                                            unsigned tile_d0, unsigned tile_d1,
                                            unsigned tens_d0, unsigned tens_d1,
                                            unsigned long long stride0) {
  const unsigned long long ga = (unsigned long long)gp;
  u32x4 g0;
  g0[0] = 1u;                                        // count=1 (valid user D#)
  g0[1] = lds_addr;                                  // LDS byte address
  g0[2] = (unsigned)(ga & 0xFFFFFFFFu);              // global_addr[31:0]
  g0[3] = (unsigned)((ga >> 32) & 0x1FFFFFFu)        // global_addr[56:32]
          | (2u << 30);                              // type = 2
  i32x8 g1;
  g1[0] = (int)(1u << 16);                           // data_size = 1 -> 2 bytes
  g1[1] = (int)((tens_d0 & 0xFFFFu) << 16);          // tensor_dim0[15:0]
  g1[2] = (int)(((tens_d0 >> 16) & 0xFFFFu) | ((tens_d1 & 0xFFFFu) << 16));
  g1[3] = (int)(((tens_d1 >> 16) & 0xFFFFu) | (tile_d0 << 16));  // tile_dim0
  g1[4] = (int)(tile_d1 & 0xFFFFu);                  // tile_dim1, tile_dim2=0
  g1[5] = (int)(stride0 & 0xFFFFFFFFull);            // tensor_dim0_stride lo
  g1[6] = (int)((stride0 >> 32) & 0xFFFFull);        // stride hi, dim1_stride=0
  g1[7] = 0;
  const i32x4 z4 = {0, 0, 0, 0};                     // 2D: groups 2/3 unused
#if __clang_major__ >= 23
  const i32x8 z8 = {0, 0, 0, 0, 0, 0, 0, 0};
  __builtin_amdgcn_tensor_load_to_lds(g0, g1, z4, z4, z8, 0);
#else
  __builtin_amdgcn_tensor_load_to_lds(g0, g1, z4, z4, 0);
#endif
}

// ---------------- bf16 WMMA GEMM: C[M,128] = A[M,K] * B[K,128] -------------
// Block tile 128x128, 8 waves as 4(M)x2(N); each wave owns 32x64 = 2x4 WMMA
// tiles, K chunked by 32, double-buffered LDS.
//  A: TDM DMA, row-major [128][32]; fragment = 2x 16B ds_load per tile.
//  B: pipelined global_load_b128 into regs during compute, then scattered to
//     fragment-major Bsw[nt][lane][e] so a fragment = one 32B ds_load.
__global__ __launch_bounds__(256)
void k_wmma_gemm(const __bf16* __restrict__ A, const __bf16* __restrict__ Bm,
                 const float* __restrict__ bias, __bf16* __restrict__ C,
                 int M, int K, int lda, int ldb,
                 long long sA, long long sB, long long sC, int relu) {
  __shared__ __attribute__((aligned(128))) __bf16 At[2][128][32];
  __shared__ __attribute__((aligned(128))) __bf16 Bsw[2][8][32][16];

  A  += (long long)blockIdx.z * sA;
  Bm += (long long)blockIdx.z * sB;
  C  += (long long)blockIdx.z * sC;

  const int blockM = blockIdx.x * 128;
  const int tid  = threadIdx.x;
  const int lane = tid & 31;
  const int wv   = tid >> 5;
  const int wm   = wv & 3;                 // M slab: 4 x 32 rows
  const int wn   = wv >> 2;                // N slab: 2 x 64 cols
  const int ncol = lane & 15;

  // B staging geometry: thread covers row brow (k in chunk), 16 cols at bn0
  const int brow = tid >> 3;               // 0..31
  const int bn0  = (tid & 7) * 16;         // 0..112
  const int bnt  = tid & 7;                // n-tile index
  const int bkh  = (brow >> 4) << 4;       // lane-group: 0 | 16
  const int be   = brow & 15;              // element index in fragment

  union BU { uint4 q[2]; __bf16 h[16]; };
  union AU { v16bf v; v8bf  h[2];  };

  v8f acc[2][4] = {};

  // ---- prologue: chunk 0 -> buffer 0
  if (tid == 0) {
    tdm_load_2d(lds_off(&At[0][0][0]), A + (size_t)blockM * lda,
                32u, 128u, (unsigned)K, (unsigned)(M - blockM),
                (unsigned long long)lda);
  }
  BU breg;
  {
    const uint4* src = reinterpret_cast<const uint4*>(Bm + (size_t)brow * ldb + bn0);
    breg.q[0] = src[0];
    breg.q[1] = src[1];
  }
#pragma unroll
  for (int j = 0; j < 16; ++j) Bsw[0][bnt][bkh + j][be] = breg.h[j];
  if (tid == 0) __builtin_amdgcn_s_wait_tensorcnt(0);
  __syncthreads();

  int buf = 0;
  for (int k0 = 0; k0 < K; k0 += 32) {
    const bool have_next = (k0 + 32) < K;
    // issue next chunk's A DMA + B global loads; both overlap the WMMAs below
    if (have_next) {
      if (tid == 0) {
        tdm_load_2d(lds_off(&At[buf ^ 1][0][0]),
                    A + (size_t)blockM * lda + (k0 + 32),
                    32u, 128u, (unsigned)(K - k0 - 32),
                    (unsigned)(M - blockM), (unsigned long long)lda);
      }
      const uint4* src =
          reinterpret_cast<const uint4*>(Bm + (size_t)(k0 + 32 + brow) * ldb + bn0);
      breg.q[0] = src[0];
      breg.q[1] = src[1];
    }

    // A fragments (16x32 bf16): lane L holds row M=L%16;
    // lanes 0-15: K = {0..15}, lanes 16-31: K = {16..31}
    const int kb = (lane < 16) ? 0 : 16;
    AU afr[2];
#pragma unroll
    for (int mt = 0; mt < 2; ++mt) {
      const int m = wm * 32 + mt * 16 + ncol;
      afr[mt].h[0] = *reinterpret_cast<const v8bf*>(&At[buf][m][kb]);
      afr[mt].h[1] = *reinterpret_cast<const v8bf*>(&At[buf][m][kb + 8]);
    }
    // B fragments: one contiguous 32B LDS read each (fragment-major layout)
#pragma unroll
    for (int nt = 0; nt < 4; ++nt) {
      const v16bf bfr =
          *reinterpret_cast<const v16bf*>(&Bsw[buf][wn * 4 + nt][lane][0]);
#pragma unroll
      for (int mt = 0; mt < 2; ++mt)
        acc[mt][nt] = __builtin_amdgcn_wmma_f32_16x16x32_bf16(
            false, afr[mt].v, false, bfr, (short)0, acc[mt][nt], false, false);
    }

    __syncthreads();                       // all waves done with buf
    if (have_next) {                       // scatter next B chunk (swizzled)
#pragma unroll
      for (int j = 0; j < 16; ++j) Bsw[buf ^ 1][bnt][bkh + j][be] = breg.h[j];
    }
    if (tid == 0) __builtin_amdgcn_s_wait_tensorcnt(0);  // next A DMA done
    __syncthreads();
    buf ^= 1;
  }

  // epilogue: C/D layout -> VGPR r: row = r (+8 for lanes 16-31), col = lane%16
#pragma unroll
  for (int mt = 0; mt < 2; ++mt) {
#pragma unroll
    for (int nt = 0; nt < 4; ++nt) {
      const int col = wn * 64 + nt * 16 + ncol;
      const float bia = bias ? bias[col] : 0.f;
#pragma unroll
      for (int r = 0; r < 8; ++r) {
        const int row = blockM + wm * 32 + mt * 16 + r + ((lane < 16) ? 0 : 8);
        float v = acc[mt][nt][r] + bia;
        if (relu) v = (v > 0.f) ? v : 0.f;
        C[(size_t)row * 128 + col] = (__bf16)v;
      }
    }
  }
}

// ---------------- pooling + classifier -------------------------------------
__global__ __launch_bounds__(128) void k_pool(const __bf16* __restrict__ h,
                                              float* __restrict__ pooled) {
  const int b = blockIdx.x, d = threadIdx.x;
  const __bf16* p = h + (size_t)b * N * Hd + d;
  float s = 0.f;
  for (int n = 0; n < N; ++n) s += (float)p[(size_t)n * Hd];
  pooled[b * Hd + d] = s * (1.f / N);
}

__global__ __launch_bounds__(128)
void k_cls(const float* __restrict__ pooled, const float* __restrict__ W1,
           const float* __restrict__ b1, const float* __restrict__ prelu,
           const float* __restrict__ W2, const float* __restrict__ b2,
           float* __restrict__ out) {
  __shared__ float x[128];
  __shared__ float t[128];
  const int b = blockIdx.x, i = threadIdx.x;
  x[i] = pooled[b * 128 + i];
  __syncthreads();
  float s = b1[i];
  for (int j = 0; j < 128; ++j) s += x[j] * W1[i * 128 + j];
  t[i] = (s > 0.f) ? s : prelu[i] * s;
  __syncthreads();
  if (i < OUTC) {
    float o = b2[i];
    for (int j = 0; j < 128; ++j) o += t[j] * W2[i * 128 + j];
    out[b * OUTC + i] = o;
  }
}

// ---------------- launch ----------------------------------------------------
extern "C" void kernel_launch(void* const* d_in, const int* in_sizes, int n_in,
                              void* d_out, int out_size, void* d_ws, size_t ws_size,
                              hipStream_t stream) {
  const float* X      = (const float*)d_in[0];
  const float* adj    = (const float*)d_in[1];
  const float* gc0_W1 = (const float*)d_in[2];
  const float* gc0_b1 = (const float*)d_in[3];
  const float* gc0_W2 = (const float*)d_in[4];
  const float* gc0_b2 = (const float*)d_in[5];
  const float* gc1_W1 = (const float*)d_in[6];
  const float* gc1_b1 = (const float*)d_in[7];
  const float* gc1_W2 = (const float*)d_in[8];
  const float* gc1_b2 = (const float*)d_in[9];
  const float* cls_W1 = (const float*)d_in[10];
  const float* cls_b1 = (const float*)d_in[11];
  const float* prelu  = (const float*)d_in[12];
  const float* cls_W2 = (const float*)d_in[13];
  const float* cls_b2 = (const float*)d_in[14];
  float* out = (float*)d_out;

  char* ws = (char*)d_ws;
  float*    S      = (float*)(ws + OFF_S);
  float*    V      = (float*)(ws + OFF_V);
  float*    deg    = (float*)(ws + OFF_DEG);
  float*    dis    = (float*)(ws + OFF_DIS);
  float*    cs     = (float*)(ws + OFF_CS);
  unsigned* bar    = (unsigned*)(ws + OFF_BAR);
  __bf16*   Ubf    = (__bf16*)(ws + OFF_UBF);
  __bf16*   Ubt    = (__bf16*)(ws + OFF_UBT);
  __bf16*   Wt0    = (__bf16*)(ws + OFF_WT0);
  __bf16*   Wt1    = (__bf16*)(ws + OFF_WT1);
  __bf16*   Wt2    = (__bf16*)(ws + OFF_WT2);
  __bf16*   Wt3    = (__bf16*)(ws + OFF_WT3);
  __bf16*   buf0   = (__bf16*)(ws + OFF_BUF0);
  __bf16*   buf1   = (__bf16*)(ws + OFF_BUF1);
  float*    pooled = (float*)(ws + OFF_POOL);

  // 1) normalized Laplacian
  k_deg <<<N, 256, 0, stream>>>(adj, deg, dis);
  k_lsym<<<N, 256, 0, stream>>>(adj, deg, dis, S);

  // 2) eigenvectors via persistent cyclic Jacobi
  k_init_bar<<<1, 1, 0, stream>>>(bar);
  k_jacobi<<<JAC_BLOCKS, 256, 0, stream>>>(S, V, cs, bar);

  // 3) bf16 conversions (U, U^T, X, MLP weights^T)
  k_cvtU<<<N, 256, 0, stream>>>(V, Ubf, Ubt);
  k_cvt <<<4096, 256, 0, stream>>>(X, buf0, (long long)Bn * N * Din);
  k_cvtWT<<<128, 128, 0, stream>>>(gc0_W1, Wt0);
  k_cvtWT<<<128, 128, 0, stream>>>(gc0_W2, Wt1);
  k_cvtWT<<<128, 128, 0, stream>>>(gc1_W1, Wt2);
  k_cvtWT<<<128, 128, 0, stream>>>(gc1_W2, Wt3);

  const long long NH = (long long)N * Hd;
  const dim3 gU(N / 128, 1, Bn);                  // per-batch U GEMMs
  const dim3 gM((Bn * N) / 128, 1, 1);            // batched-row MLP GEMMs
  const dim3 blk(256, 1, 1);

  // GraphConv 0: relu( U @ MLP( U^T @ X ) )
  k_wmma_gemm<<<gU, blk, 0, stream>>>(Ubt, buf0, nullptr, buf1, N, N, N, 128, 0, NH, NH, 0);
  k_wmma_gemm<<<gM, blk, 0, stream>>>(buf1, Wt0, gc0_b1, buf0, Bn * N, 128, 128, 128, 0, 0, 0, 1);
  k_wmma_gemm<<<gM, blk, 0, stream>>>(buf0, Wt1, gc0_b2, buf1, Bn * N, 128, 128, 128, 0, 0, 0, 0);
  k_wmma_gemm<<<gU, blk, 0, stream>>>(Ubf, buf1, nullptr, buf0, N, N, N, 128, 0, NH, NH, 1);

  // GraphConv 1
  k_wmma_gemm<<<gU, blk, 0, stream>>>(Ubt, buf0, nullptr, buf1, N, N, N, 128, 0, NH, NH, 0);
  k_wmma_gemm<<<gM, blk, 0, stream>>>(buf1, Wt2, gc1_b1, buf0, Bn * N, 128, 128, 128, 0, 0, 0, 1);
  k_wmma_gemm<<<gM, blk, 0, stream>>>(buf0, Wt3, gc1_b2, buf1, Bn * N, 128, 128, 128, 0, 0, 0, 0);
  k_wmma_gemm<<<gU, blk, 0, stream>>>(Ubf, buf1, nullptr, buf0, N, N, N, 128, 0, NH, NH, 1);

  // 4) mean pool + classifier
  k_pool<<<Bn, 128, 0, stream>>>(buf0, pooled);
  k_cls <<<Bn, 128, 0, stream>>>(pooled, cls_W1, cls_b1, prelu, cls_W2, cls_b2, out);
}
